// EncoderLayer_86741159510211
// MI455X (gfx1250) — compile-verified, matrix operand order checked by reading, not run on hardware
//
#include <hip/hip_runtime.h>
#include <hip/hip_bf16.h>
#include <math.h>
#include <stdint.h>

// ---------------------------------------------------------------------------
// Types
// ---------------------------------------------------------------------------
typedef __attribute__((ext_vector_type(16))) _Float16 v16h;
typedef __attribute__((ext_vector_type(8)))  _Float16 v8h;
typedef __attribute__((ext_vector_type(2)))  _Float16 h2v;
typedef __attribute__((ext_vector_type(8)))  float    v8f;
// gcc-style vector to match the async-LDS builtin's parameter type exactly
typedef int v4i_t __attribute__((vector_size(16)));

__device__ __forceinline__ int lane_id() { return (int)(threadIdx.x & 31); }

// ---------------------------------------------------------------------------
// CDNA5 async global->LDS copy (ASYNCcnt-tracked, no VGPR round trip).
// Builtin signature (from the round-2 diagnostic): 
//   void __builtin_amdgcn_global_load_async_to_lds_b128(
//       v4i addrspace(1)* gaddr, v4i addrspace(3)* lds,
//       constant int offset, constant int cpol)
// ---------------------------------------------------------------------------
#if defined(__has_builtin)
#if __has_builtin(__builtin_amdgcn_global_load_async_to_lds_b128)
#define HAVE_ASYNC_LDS 1
#endif
#endif
#ifndef HAVE_ASYNC_LDS
#define HAVE_ASYNC_LDS 0
#endif

__device__ __forceinline__ void async_b128(const _Float16* g, _Float16* l) {
#if HAVE_ASYNC_LDS
  __builtin_amdgcn_global_load_async_to_lds_b128(
      (__attribute__((address_space(1))) v4i_t*)(uintptr_t)g,
      (__attribute__((address_space(3))) v4i_t*)(uint32_t)(uintptr_t)l, 0, 0);
#else
  *(v8h*)l = *(const v8h*)g;
#endif
}

__device__ __forceinline__ void wait_async() {
#if HAVE_ASYNC_LDS
#if __has_builtin(__builtin_amdgcn_s_wait_asynccnt)
  __builtin_amdgcn_s_wait_asynccnt(0);
#else
  asm volatile("s_wait_asynccnt 0x0" ::: "memory");
#endif
#endif
}

// ---------------------------------------------------------------------------
// WMMA fragment gathers (ISA 7.12.2 layouts, wave32)
// A fragment: 16x32 f16, element (m,k) at src[m*ld + k]
//   lane: m = lane&15, g = lane>>4;  vgpr j holds K = base(j) + {0,1} + g*8
//   base(j) = (j&4 ? 16 : 0) + (j&3)*2
// ---------------------------------------------------------------------------
__device__ __forceinline__ v16h load_frag_a(const _Float16* src, int ld) {
  const int lane = lane_id();
  const int m = lane & 15;
  const int g = lane >> 4;
  v16h f;
#pragma unroll
  for (int j = 0; j < 8; ++j) {
    const int k = ((j & 4) ? 16 : 0) + (j & 3) * 2 + g * 8;
    h2v p = *(const h2v*)(src + (size_t)m * ld + k);
    f[2 * j]     = p[0];
    f[2 * j + 1] = p[1];
  }
  return f;
}

// B fragment from K-contiguous (transposed) source: element (k,n) at
// src[n*ld + k].  Per-lane reads are contiguous in K -> vectorizes to
// ds_load_b128 pairs.
__device__ __forceinline__ v16h load_frag_bt(const _Float16* src, int ld) {
  const int lane = lane_id();
  const int n = lane & 15;
  const int g = lane >> 4;
  v16h f;
#pragma unroll
  for (int j = 0; j < 8; ++j) {
    const int k = ((j & 4) ? 16 : 0) + (j & 3) * 2 + g * 8;
    h2v p = *(const h2v*)(src + (size_t)n * ld + k);
    f[2 * j]     = p[0];
    f[2 * j + 1] = p[1];
  }
  return f;
}

__device__ __forceinline__ v8f wmma_f16(v16h a, v16h b, v8f c) {
  return __builtin_amdgcn_wmma_f32_16x16x32_f16(false, a, false, b, (short)0, c,
                                                false, false);
}

// ---------------------------------------------------------------------------
// fp32 [K][N] -> f16 [N][K] transposing convert (LDS-tiled, coalesced)
// ---------------------------------------------------------------------------
__global__ void __launch_bounds__(256)
cvt_transpose_f32_f16(const float* __restrict__ in, _Float16* __restrict__ out,
                      int K, int N) {
  __shared__ float tile[32][33];
  const int nb = blockIdx.x * 32;
  const int kb = blockIdx.y * 32;
  const int tx = threadIdx.x & 31;
  const int ty = threadIdx.x >> 5;  // 8 row-groups
  for (int r = ty; r < 32; r += 8)
    tile[r][tx] = in[(size_t)(kb + r) * N + nb + tx];
  __syncthreads();
  for (int r = ty; r < 32; r += 8)
    out[(size_t)(nb + r) * K + kb + tx] = (_Float16)tile[tx][r];
}

// ---------------------------------------------------------------------------
// LayerNorm (torch style: g*(x-mean)/(std+eps)+b, std unbiased ddof=1) -> f16
// ---------------------------------------------------------------------------
__global__ void __launch_bounds__(256)
layernorm_f16(const float* __restrict__ x, const float* __restrict__ g,
              const float* __restrict__ b, _Float16* __restrict__ out, int D) {
  const int row = blockIdx.x;
  const float* xr = x + (size_t)row * D;
  __shared__ float red[256];

  float s = 0.f;
  for (int i = threadIdx.x; i < D; i += 256) s += xr[i];
  red[threadIdx.x] = s;
  __syncthreads();
  for (int o = 128; o > 0; o >>= 1) {
    if ((int)threadIdx.x < o) red[threadIdx.x] += red[threadIdx.x + o];
    __syncthreads();
  }
  const float mean = red[0] / (float)D;
  __syncthreads();

  float s2 = 0.f;
  for (int i = threadIdx.x; i < D; i += 256) {
    float d = xr[i] - mean;
    s2 += d * d;
  }
  red[threadIdx.x] = s2;
  __syncthreads();
  for (int o = 128; o > 0; o >>= 1) {
    if ((int)threadIdx.x < o) red[threadIdx.x] += red[threadIdx.x + o];
    __syncthreads();
  }
  const float var = red[0] / (float)(D - 1);
  const float inv = 1.0f / (sqrtf(var) + 1e-6f);

  for (int i = threadIdx.x; i < D; i += 256)
    out[(size_t)row * D + i] = (_Float16)(g[i] * (xr[i] - mean) * inv + b[i]);
}

// ---------------------------------------------------------------------------
// Tiled WMMA GEMM:  out[M,N] = act( A[M,K](f16) @ W + bias ) (+resid)
// Weights pre-transposed: Bt[N][K] f16  ->  all B fragments are K-contiguous.
// Workgroup: 256 threads = 8 waves; tile 64(M) x 128(N), K-step 32.
// Wave w -> rows (w&3)*16, cols (w>>2)*64 : 4 accumulators of 16x16.
// Global->LDS staging via async b128 copies (ASYNCcnt) when available.
// ---------------------------------------------------------------------------
constexpr int TM = 64, TN = 128, TK = 32;
constexpr int LDT = TK + 16;  // 48 halves: keeps 16B alignment for b128 ops

template <typename OutT, bool RELU, bool RES>
__global__ void __launch_bounds__(256)
gemm_wmma(const _Float16* __restrict__ A, const _Float16* __restrict__ Bt,
          const float* __restrict__ bias, const float* resid, OutT* out,
          int M, int N, int K) {
  __shared__ _Float16 As[TM * LDT];  // 64 rows(m) x 32(k)
  __shared__ _Float16 Bs[TN * LDT];  // 128 rows(n) x 32(k)
  (void)M;

  const int n0 = blockIdx.x * TN;
  const int m0 = blockIdx.y * TM;
  const int t = threadIdx.x;
  const int wave = t >> 5;
  const int wm = (wave & 3) * 16;
  const int wn = (wave >> 2) * 64;

  v8f acc[4] = {};

  const int arow = t >> 2, acol = (t & 3) * 8;   // A tile: 1 b128 / thread
  const int brow = t >> 1, bcol = (t & 1) * 16;  // B tile: 2 b128 / thread

  const _Float16* ag = &A[(size_t)(m0 + arow) * K + acol];
  const _Float16* bg = &Bt[(size_t)(n0 + brow) * K + bcol];

  for (int k0 = 0; k0 < K; k0 += TK) {
    async_b128(ag + k0, &As[arow * LDT + acol]);
    async_b128(bg + k0, &Bs[brow * LDT + bcol]);
    async_b128(bg + k0 + 8, &Bs[brow * LDT + bcol + 8]);
    if (k0 + TK < K) {  // pull next K-tile toward L2/WGP$
      __builtin_prefetch(ag + k0 + TK, 0, 0);
      __builtin_prefetch(bg + k0 + TK, 0, 0);
    }
    wait_async();
    __syncthreads();

    v16h af = load_frag_a(&As[wm * LDT], LDT);
#pragma unroll
    for (int nt = 0; nt < 4; ++nt) {
      v16h bf = load_frag_bt(&Bs[(wn + nt * 16) * LDT], LDT);
      acc[nt] = wmma_f16(af, bf, acc[nt]);
    }
    __syncthreads();
  }

  const int lane = t & 31;
  const int cn = lane & 15;
  const int cg = lane >> 4;
#pragma unroll
  for (int nt = 0; nt < 4; ++nt) {
    const int col = n0 + wn + nt * 16 + cn;
    const float bv = bias[col];
#pragma unroll
    for (int r = 0; r < 8; ++r) {
      const int row = m0 + wm + r + 8 * cg;
      float v = acc[nt][r] + bv;
      if (RELU) v = v > 0.0f ? v : 0.0f;
      if (RES) v += resid[(size_t)row * N + col];
      out[(size_t)row * N + col] = (OutT)v;
    }
  }
}

// ---------------------------------------------------------------------------
// Flash attention: one WG (4 waves / 128 threads) handles 64 query rows of
// one (batch, head); iterates 64-key blocks with online softmax.
// Q/K/V/ctx layout: [b*S + t][h*64 + d]  (token-major, heads packed in D)
// K tile staged row-major (Q.K^T gathers are K-contiguous); V tile staged
// TRANSPOSED in LDS so P.V B-fragments are K-contiguous too.
// ---------------------------------------------------------------------------
constexpr int AQ = 64, AK = 64;
constexpr int LDS_T = 64 + 8;  // 72 halves

__global__ void __launch_bounds__(128)
flash_attn(const _Float16* __restrict__ Q, const _Float16* __restrict__ Kk,
           const _Float16* __restrict__ V, const int* __restrict__ mask,
           _Float16* __restrict__ ctx, int S, int D, int H) {
  __shared__ _Float16 Kt[AK * LDS_T];   // [key][dk]
  __shared__ _Float16 VtT[AK * LDS_T];  // [dk][key]  (transposed)
  __shared__ _Float16 Pt[AQ * LDS_T];   // [q][key]   (also Q staging)

  const int bh = blockIdx.x;
  const int b = bh / H, h = bh % H;
  const int q0 = blockIdx.y * AQ;
  const int t = threadIdx.x;
  const int wave = t >> 5;
  const int lane = t & 31;
  const int cn = lane & 15;
  const int cg = lane >> 4;
  const size_t base = (size_t)b * S * D + (size_t)h * 64;

  const int lrow = t >> 1, lcb = (t & 1) * 32;  // 64 rows x 2 half-rows

  // stage Q tile through Pt (async), pull A fragments into registers
#pragma unroll
  for (int j = 0; j < 4; ++j)
    async_b128(&Q[base + (size_t)(q0 + lrow) * D + lcb + j * 8],
               &Pt[lrow * LDS_T + lcb + j * 8]);
  wait_async();
  __syncthreads();
  v16h aq0 = load_frag_a(&Pt[(wave * 16) * LDS_T], LDS_T);
  v16h aq1 = load_frag_a(&Pt[(wave * 16) * LDS_T + 32], LDS_T);
  __syncthreads();

  float mrow[8], lsum[8];
#pragma unroll
  for (int r = 0; r < 8; ++r) { mrow[r] = -3.0e38f; lsum[r] = 0.f; }
  v8f oacc[4] = {};

  const float scale = 0.125f;  // 1/sqrt(64)
  const int nblocks = S / AK;

  for (int kb = 0; kb < nblocks; ++kb) {
    const int key0 = kb * AK;
    // K tile: async row-major copy
#pragma unroll
    for (int j = 0; j < 4; ++j)
      async_b128(&Kk[base + (size_t)(key0 + lrow) * D + lcb + j * 8],
                 &Kt[lrow * LDS_T + lcb + j * 8]);
    // V tile: load through VGPRs and store transposed ([dk][key])
#pragma unroll
    for (int j = 0; j < 4; ++j) {
      v8h vv = *(const v8h*)&V[base + (size_t)(key0 + lrow) * D + lcb + j * 8];
#pragma unroll
      for (int e = 0; e < 8; ++e)
        VtT[(lcb + j * 8 + e) * LDS_T + lrow] = vv[e];
    }
    wait_async();
    __syncthreads();

    // S = Q K^T  (16 query rows x 64 keys per wave)
    v8f sacc[4] = {};
#pragma unroll
    for (int nt = 0; nt < 4; ++nt) {
      v16h b0 = load_frag_bt(&Kt[(nt * 16) * LDS_T + 0], LDS_T);
      sacc[nt] = wmma_f16(aq0, b0, sacc[nt]);
      v16h b1 = load_frag_bt(&Kt[(nt * 16) * LDS_T + 32], LDS_T);
      sacc[nt] = wmma_f16(aq1, b1, sacc[nt]);
    }

    // scale + mask + per-row max
    float rmax[8];
#pragma unroll
    for (int r = 0; r < 8; ++r) rmax[r] = -3.0e38f;
#pragma unroll
    for (int nt = 0; nt < 4; ++nt) {
      const int key = key0 + nt * 16 + cn;
      const bool mz = (mask[(size_t)b * S + key] == 0);
#pragma unroll
      for (int r = 0; r < 8; ++r) {
        float sv = sacc[nt][r] * scale;
        if (mz) sv = -1.0e9f;
        sacc[nt][r] = sv;
        rmax[r] = fmaxf(rmax[r], sv);
      }
    }
    // reduce across the 16 lanes sharing a row (xor of bits<4 stays in half)
#pragma unroll
    for (int r = 0; r < 8; ++r) {
#pragma unroll
      for (int off = 8; off > 0; off >>= 1)
        rmax[r] = fmaxf(rmax[r], __shfl_xor(rmax[r], off, 32));
    }

    float alpha[8], rsum[8];
#pragma unroll
    for (int r = 0; r < 8; ++r) {
      const float mnew = fmaxf(mrow[r], rmax[r]);
      alpha[r] = __expf(mrow[r] - mnew);
      mrow[r] = mnew;
      rsum[r] = 0.f;
    }
#pragma unroll
    for (int nt = 0; nt < 4; ++nt) {
#pragma unroll
      for (int r = 0; r < 8; ++r) {
        float p = __expf(sacc[nt][r] - mrow[r]);
        sacc[nt][r] = p;
        rsum[r] += p;
      }
    }
#pragma unroll
    for (int r = 0; r < 8; ++r) {
#pragma unroll
      for (int off = 8; off > 0; off >>= 1)
        rsum[r] += __shfl_xor(rsum[r], off, 32);
      lsum[r] = lsum[r] * alpha[r] + rsum[r];
    }

    // rescale running O, spill P to LDS (C-layout -> memory)
#pragma unroll
    for (int nt = 0; nt < 4; ++nt) {
#pragma unroll
      for (int r = 0; r < 8; ++r) {
        oacc[nt][r] *= alpha[r];
        Pt[(wave * 16 + r + 8 * cg) * LDS_T + nt * 16 + cn] =
            (_Float16)sacc[nt][r];
      }
    }
    __syncthreads();

    // O += P V   (B fragments K-contiguous from transposed V tile)
#pragma unroll
    for (int kc = 0; kc < 2; ++kc) {
      v16h ap = load_frag_a(&Pt[(wave * 16) * LDS_T + kc * 32], LDS_T);
#pragma unroll
      for (int nt = 0; nt < 4; ++nt) {
        v16h bv = load_frag_bt(&VtT[(nt * 16) * LDS_T + kc * 32], LDS_T);
        oacc[nt] = wmma_f16(ap, bv, oacc[nt]);
      }
    }
    __syncthreads();
  }

  // epilogue: O / l  -> ctx (f16)
#pragma unroll
  for (int nt = 0; nt < 4; ++nt) {
#pragma unroll
    for (int r = 0; r < 8; ++r) {
      const int row = q0 + wave * 16 + r + 8 * cg;
      ctx[base + (size_t)row * D + nt * 16 + cn] =
          (_Float16)(oacc[nt][r] / lsum[r]);
    }
  }
}

// ---------------------------------------------------------------------------
// Host-side orchestration
// ---------------------------------------------------------------------------
extern "C" void kernel_launch(void* const* d_in, const int* in_sizes, int n_in,
                              void* d_out, int out_size, void* d_ws, size_t ws_size,
                              hipStream_t stream) {
  (void)in_sizes; (void)n_in; (void)out_size; (void)ws_size;

  const float* x     = (const float*)d_in[0];
  const int*   mask  = (const int*)d_in[1];
  const float* wq    = (const float*)d_in[2];
  const float* wk    = (const float*)d_in[3];
  const float* wv    = (const float*)d_in[4];
  const float* wo    = (const float*)d_in[5];
  const float* bq    = (const float*)d_in[6];
  const float* bk    = (const float*)d_in[7];
  const float* bv    = (const float*)d_in[8];
  const float* bo    = (const float*)d_in[9];
  const float* w1    = (const float*)d_in[10];
  const float* b1    = (const float*)d_in[11];
  const float* w2    = (const float*)d_in[12];
  const float* b2    = (const float*)d_in[13];
  const float* ln1_g = (const float*)d_in[14];
  const float* ln1_b = (const float*)d_in[15];
  const float* ln2_g = (const float*)d_in[16];
  const float* ln2_b = (const float*)d_in[17];
  float* out = (float*)d_out;

  constexpr int B = 2, S = 2048, D = 1024, H = 16, FF = 4096;
  const int T = B * S;

  char* ws = (char*)d_ws;
  size_t off = 0;
  auto alloc = [&](size_t bytes) {
    size_t o = off;
    off += (bytes + 255) & ~(size_t)255;
    return o;
  };
  // transposed f16 weights: [N][K]
  _Float16* wq16 = (_Float16*)(ws + alloc((size_t)D * D * 2));
  _Float16* wk16 = (_Float16*)(ws + alloc((size_t)D * D * 2));
  _Float16* wv16 = (_Float16*)(ws + alloc((size_t)D * D * 2));
  _Float16* wo16 = (_Float16*)(ws + alloc((size_t)D * D * 2));
  _Float16* w116 = (_Float16*)(ws + alloc((size_t)D * FF * 2));  // [FF][D]
  _Float16* w216 = (_Float16*)(ws + alloc((size_t)FF * D * 2));  // [D][FF]
  _Float16* h16  = (_Float16*)(ws + alloc((size_t)T * D * 2));
  _Float16* q16  = (_Float16*)(ws + alloc((size_t)T * D * 2));
  _Float16* k16  = (_Float16*)(ws + alloc((size_t)T * D * 2));
  _Float16* v16  = (_Float16*)(ws + alloc((size_t)T * D * 2));
  _Float16* c16  = (_Float16*)(ws + alloc((size_t)T * D * 2));
  _Float16* ff16 = (_Float16*)(ws + alloc((size_t)T * FF * 2));

  dim3 tb(256);
  auto cvtT = [&](const float* src, _Float16* dst, int K, int N) {
    cvt_transpose_f32_f16<<<dim3(N / 32, K / 32), tb, 0, stream>>>(src, dst, K, N);
  };
  cvtT(wq, wq16, D, D);
  cvtT(wk, wk16, D, D);
  cvtT(wv, wv16, D, D);
  cvtT(wo, wo16, D, D);
  cvtT(w1, w116, D, FF);
  cvtT(w2, w216, FF, D);

  // sublayer 1: LN1 -> QKV -> flash attention -> Wo + residual
  layernorm_f16<<<dim3(T), tb, 0, stream>>>(x, ln1_g, ln1_b, h16, D);

  dim3 gD(D / TN, T / TM);  // N=1024
  gemm_wmma<_Float16, false, false><<<gD, tb, 0, stream>>>(h16, wq16, bq, nullptr, q16, T, D, D);
  gemm_wmma<_Float16, false, false><<<gD, tb, 0, stream>>>(h16, wk16, bk, nullptr, k16, T, D, D);
  gemm_wmma<_Float16, false, false><<<gD, tb, 0, stream>>>(h16, wv16, bv, nullptr, v16, T, D, D);

  dim3 gA(B * H, S / AQ);
  flash_attn<<<gA, dim3(128), 0, stream>>>(q16, k16, v16, mask, c16, S, D, H);

  gemm_wmma<float, false, true><<<gD, tb, 0, stream>>>(c16, wo16, bo, x, out, T, D, D);

  // sublayer 2: LN2 -> FFN (ReLU) -> residual (in-place on d_out)
  layernorm_f16<<<dim3(T), tb, 0, stream>>>(out, ln2_g, ln2_b, h16, D);

  dim3 gF(FF / TN, T / TM);  // N=4096
  gemm_wmma<_Float16, true, false><<<gF, tb, 0, stream>>>(h16, w116, b1, nullptr, ff16, T, FF, D);
  gemm_wmma<float, false, true><<<gD, tb, 0, stream>>>(ff16, w216, b2, out, out, T, D, FF);
}
